// EdgeDecoder_85461259256120
// MI455X (gfx1250) — matrix operand order repaired and checked by reading, not compiled, not run on hardware
//
#include <hip/hip_runtime.h>
#include <hip/hip_bf16.h>

typedef __attribute__((ext_vector_type(16))) __bf16 v16bf;
typedef __attribute__((ext_vector_type(8)))  float  v8f;
typedef __attribute__((ext_vector_type(4)))  __bf16 bf16x4;

#define EMB      256
#define TILE_E   64      // edges per block (2 blocks co-resident per WGP)
#define THREADS  256     // 8 waves (wave32)
#define MTILES   (TILE_E / 16)   // 4

// padded LDS strides (elements) to skew banks
#define ZS   520   // [64][520] bf16  -> 66560 B
#define H1S  264   // [64][264] bf16  -> 33792 B
#define H2S  132   // [64][132] f32   -> 33792 B
#define LDS_OFF_H1  66560
#define LDS_OFF_H2  (66560 + 33792)
#define LDS_BYTES   (66560 + 33792 + 33792)   // 134144 B -> 2 blocks / 320KB WGP

// W1 fragments: 16 ntiles x 16 ksteps x (32 lanes x 16 bf16) = 131072 elems
// W2 fragments:  8 ntiles x  8 ksteps x (32 lanes x 16 bf16) =  32768 elems
#define W1P_ELEMS 131072
#define W2P_ELEMS 32768

// ---------------------------------------------------------------------------
// Prepack f32 weights into bf16 B fragments in this code's contiguous K order:
//   B (32x16): lanes 0-15 hold K=0..15 (N=lane), lanes 16-31 hold K=16..31,
//   each lane: 16 contiguous K values (2 per VGPR). Stored lane-contiguous so
//   the main loop does two global_load_b128 per fragment.
// ---------------------------------------------------------------------------
__global__ void prepack_weights(const float* __restrict__ W1,
                                const float* __restrict__ W2,
                                __hip_bfloat16* __restrict__ w1p,
                                __hip_bfloat16* __restrict__ w2p) {
    int idx = blockIdx.x * blockDim.x + threadIdx.x;
    if (idx < W1P_ELEMS) {
        int f = idx >> 9, within = idx & 511;
        int lane = within >> 4, e = within & 15;
        int ntile = f >> 4, kstep = f & 15;          // W1: 16 ntiles x 16 ksteps
        int n = lane & 15, khalf = lane >> 4;
        int K = kstep * 32 + khalf * 16 + e;         // 0..511
        int N = ntile * 16 + n;                      // 0..255
        w1p[idx] = __float2bfloat16(W1[K * 256 + N]);
    } else if (idx < W1P_ELEMS + W2P_ELEMS) {
        int j = idx - W1P_ELEMS;
        int f = j >> 9, within = j & 511;
        int lane = within >> 4, e = within & 15;
        int ntile = f >> 3, kstep = f & 7;           // W2: 8 ntiles x 8 ksteps
        int n = lane & 15, khalf = lane >> 4;
        int K = kstep * 32 + khalf * 16 + e;         // 0..255
        int N = ntile * 16 + n;                      // 0..127
        w2p[j] = __float2bfloat16(W2[K * 128 + N]);
    }
}

// B / A fragment container
struct BF { union { v16bf v; uint4 u[2]; }; };

// B fragment: lane-contiguous 32B in prepacked buffer
__device__ __forceinline__ void loadB(BF& b, const __bf16* base, int nt, int ks,
                                      int lane, int nstride) {
    const __bf16* p = base + ((size_t)(nt * nstride + ks) * 32 + lane) * 16;
    b.u[0] = *(const uint4*)(p);
    b.u[1] = *(const uint4*)(p + 8);
}

// A fragment from LDS (contiguous K order, same as packed B): lane<16 -> K {0..15},
// lane>=16 -> K {16..31} ; two 16B chunks per lane.
__device__ __forceinline__ void loadA(BF& a, const __bf16* base, int rstride,
                                      int m0, int nlo, int ks, int khi8) {
    const __bf16* p = base + (size_t)(m0 + nlo) * rstride + ks * 32 + khi8 * 2;
    a.u[0] = *(const uint4*)(p);
    a.u[1] = *(const uint4*)(p + 8);
}

#define WMMA_BF16(Af, Bf, Cf) \
    __builtin_amdgcn_wmma_f32_16x16x32_bf16(false, (Af), false, (Bf), (short)0, (Cf), false, false)

// ---------------------------------------------------------------------------
// Fused edge decoder: gather -> (512x256) -> ReLU -> (256x128) -> ReLU ->
// (128x1) -> sigmoid.  One block = 64 edges; each wave owns an N-slice and
// sweeps all 4 M-tiles; two blocks per WGP so gather and WMMA phases of
// different blocks overlap.
// ---------------------------------------------------------------------------
__global__ __launch_bounds__(THREADS)
void edge_decoder(const float* __restrict__ zs, const float* __restrict__ zo,
                  const int* __restrict__ row, const int* __restrict__ col,
                  const __hip_bfloat16* __restrict__ w1p_,
                  const float* __restrict__ b1,
                  const __hip_bfloat16* __restrict__ w2p_,
                  const float* __restrict__ b2,
                  const float* __restrict__ W3, const float* __restrict__ b3,
                  float* __restrict__ out, int E) {
    extern __shared__ char smem[];
    __bf16* Zl  = (__bf16*)smem;                  // [64][ZS]
    __bf16* H1l = (__bf16*)(smem + LDS_OFF_H1);   // [64][H1S]
    float*  H2l = (float*)(smem + LDS_OFF_H2);    // [64][H2S]
    const __bf16* w1p = (const __bf16*)w1p_;
    const __bf16* w2p = (const __bf16*)w2p_;

    const int tid   = threadIdx.x;
    const int ebase = blockIdx.x * TILE_E;

    // ---- stage gathered embeddings into LDS as bf16 (4 threads / edge) ----
    {
        int eloc = tid >> 2;                 // 0..63
        int q    = tid & 3;                  // quarter of the 512-dim concat
        int eg = ebase + eloc; if (eg >= E) eg = E - 1;
        int idx = (q < 2) ? row[eg] : col[eg];
        const float* src = ((q < 2) ? zs : zo) + (size_t)idx * EMB + (q & 1) * 128;
        __bf16* dst = Zl + (size_t)eloc * ZS + q * 128;
        #pragma unroll 8
        for (int k = 0; k < 128; k += 4) {
            float4 f = *(const float4*)(src + k);
            bf16x4 v;
            v[0] = (__bf16)f.x; v[1] = (__bf16)f.y;
            v[2] = (__bf16)f.z; v[3] = (__bf16)f.w;
            *(bf16x4*)(dst + k) = v;
        }
    }
    __syncthreads();

    const int lane = tid & 31;
    const int wave = tid >> 5;           // 0..7
    const int nlo  = lane & 15;          // N column / A-row M for this lane
    const int khi8 = (lane >> 4) * 8;    // A-layout K offset & D-layout M offset

    // ----- layer 1: [64x512] @ [512x256]; wave owns ntiles {2w, 2w+1} ------
    {
        const int n0t = wave * 2;
        v8f acc[MTILES][2];
        float bv0 = b1[n0t * 16 + nlo];
        float bv1 = b1[(n0t + 1) * 16 + nlo];
        #pragma unroll
        for (int mt = 0; mt < MTILES; ++mt)
            #pragma unroll
            for (int r = 0; r < 8; ++r) { acc[mt][0][r] = bv0; acc[mt][1][r] = bv1; }

        BF Bc0, Bc1, Bn0, Bn1, A;
        loadB(Bc0, w1p, n0t,     0, lane, 16);
        loadB(Bc1, w1p, n0t + 1, 0, lane, 16);
        #pragma unroll 1
        for (int kp = 0; kp < 8; ++kp) {
            const int ks0 = kp * 2, ks1 = ks0 + 1;
            // prefetch odd k-step while computing even
            loadB(Bn0, w1p, n0t,     ks1, lane, 16);
            loadB(Bn1, w1p, n0t + 1, ks1, lane, 16);
            #pragma unroll
            for (int mt = 0; mt < MTILES; ++mt) {
                loadA(A, Zl, ZS, mt * 16, nlo, ks0, khi8);
                acc[mt][0] = WMMA_BF16(A.v, Bc0.v, acc[mt][0]);
                acc[mt][1] = WMMA_BF16(A.v, Bc1.v, acc[mt][1]);
            }
            const int ks2 = (kp < 7) ? ks0 + 2 : ks0;  // clamped (dead) prefetch
            loadB(Bc0, w1p, n0t,     ks2, lane, 16);
            loadB(Bc1, w1p, n0t + 1, ks2, lane, 16);
            #pragma unroll
            for (int mt = 0; mt < MTILES; ++mt) {
                loadA(A, Zl, ZS, mt * 16, nlo, ks1, khi8);
                acc[mt][0] = WMMA_BF16(A.v, Bn0.v, acc[mt][0]);
                acc[mt][1] = WMMA_BF16(A.v, Bn1.v, acc[mt][1]);
            }
        }
        // ReLU -> bf16 -> LDS (D layout: VGPR r, lanes<16 M=r, lanes>=16 M=r+8)
        #pragma unroll
        for (int mt = 0; mt < MTILES; ++mt)
            #pragma unroll
            for (int j = 0; j < 2; ++j)
                #pragma unroll
                for (int r = 0; r < 8; ++r) {
                    float v = acc[mt][j][r]; v = v > 0.f ? v : 0.f;
                    H1l[(size_t)(mt * 16 + r + khi8) * H1S + (n0t + j) * 16 + nlo] = (__bf16)v;
                }
    }
    __syncthreads();

    // ----- layer 2: [64x256] @ [256x128]; wave owns ntile w ----------------
    {
        v8f acc[MTILES];
        float bv = b2[wave * 16 + nlo];
        #pragma unroll
        for (int mt = 0; mt < MTILES; ++mt)
            #pragma unroll
            for (int r = 0; r < 8; ++r) acc[mt][r] = bv;

        BF Bc, Bn, A;
        loadB(Bc, w2p, wave, 0, lane, 8);
        #pragma unroll 1
        for (int kp = 0; kp < 4; ++kp) {
            const int ks0 = kp * 2, ks1 = ks0 + 1;
            loadB(Bn, w2p, wave, ks1, lane, 8);
            #pragma unroll
            for (int mt = 0; mt < MTILES; ++mt) {
                loadA(A, H1l, H1S, mt * 16, nlo, ks0, khi8);
                acc[mt] = WMMA_BF16(A.v, Bc.v, acc[mt]);
            }
            const int ks2 = (kp < 3) ? ks0 + 2 : ks0;
            loadB(Bc, w2p, wave, ks2, lane, 8);
            #pragma unroll
            for (int mt = 0; mt < MTILES; ++mt) {
                loadA(A, H1l, H1S, mt * 16, nlo, ks1, khi8);
                acc[mt] = WMMA_BF16(A.v, Bn.v, acc[mt]);
            }
        }
        #pragma unroll
        for (int mt = 0; mt < MTILES; ++mt)
            #pragma unroll
            for (int r = 0; r < 8; ++r) {
                float v = acc[mt][r]; v = v > 0.f ? v : 0.f;
                H2l[(size_t)(mt * 16 + r + khi8) * H2S + wave * 16 + nlo] = v;
            }
    }
    __syncthreads();

    // -------------------- layer 3: [128] . [128] -> sigmoid -----------------
    if (tid < TILE_E) {
        int eg = ebase + tid;
        const float* h = H2l + (size_t)tid * H2S;
        float sum = b3[0];
        #pragma unroll 8
        for (int k = 0; k < 128; k += 4) {
            float4 hv = *(const float4*)(h + k);
            float4 wv = *(const float4*)(W3 + k);
            sum += hv.x * wv.x + hv.y * wv.y + hv.z * wv.z + hv.w * wv.w;
        }
        float s = 1.0f / (1.0f + __expf(-sum));
        if (eg < E) out[eg] = s;
    }
}

// ---------------------------------------------------------------------------
extern "C" void kernel_launch(void* const* d_in, const int* in_sizes, int n_in,
                              void* d_out, int out_size, void* d_ws, size_t ws_size,
                              hipStream_t stream) {
    const float* zs = (const float*)d_in[0];
    const float* zo = (const float*)d_in[1];
    const int*   row = (const int*)d_in[2];
    const int*   col = (const int*)d_in[3];
    const float* W1 = (const float*)d_in[4];
    const float* b1 = (const float*)d_in[5];
    const float* W2 = (const float*)d_in[6];
    const float* b2 = (const float*)d_in[7];
    const float* W3 = (const float*)d_in[8];
    const float* b3 = (const float*)d_in[9];
    const int E = in_sizes[2];

    __hip_bfloat16* w1p = (__hip_bfloat16*)d_ws;
    __hip_bfloat16* w2p = w1p + W1P_ELEMS;

    const int prep_total = W1P_ELEMS + W2P_ELEMS;
    prepack_weights<<<(prep_total + 255) / 256, 256, 0, stream>>>(W1, W2, w1p, w2p);

    const int nblk = (E + TILE_E - 1) / TILE_E;
    edge_decoder<<<nblk, THREADS, LDS_BYTES, stream>>>(
        zs, zo, row, col, w1p, b1, w2p, b2, W3, b3, (float*)d_out, E);
}
